// DCN_80187039416369
// MI455X (gfx1250) — matrix operand order repaired and checked by reading, not compiled, not run on hardware
//
#include <hip/hip_runtime.h>
#include <hip/hip_bf16.h>
#include <stdint.h>

// ---------------------------------------------------------------------------
// Deformable conv forward for MI455X (gfx1250, wave32, WMMA).
//   Kernel 1: offset conv (64 -> 18 ch, 3x3, pad 1)  -> d_ws
//   Kernel 2: bilinear sampling -> bf16 A fragments in LDS,
//             weight -> bf16 B fragments in LDS,
//             v_wmma_f32_16x16x32_bf16 implicit GEMM (M=pixels, N=64, K=576)
// ---------------------------------------------------------------------------

typedef __bf16 bf16_t;
typedef bf16_t v16bf __attribute__((ext_vector_type(16)));
typedef float  v8f   __attribute__((ext_vector_type(8)));

#define Bn  4
#define Cn  64
#define Hn  128
#define Wn  128
#define On  64
#define HWn (Hn * Wn)
#define KK  576   // C * 9

__device__ __forceinline__ unsigned short f2bf(float f) {
  // round-to-nearest-even fp32 -> bf16
  unsigned u = __float_as_uint(f);
  u += 0x7FFFu + ((u >> 16) & 1u);
  return (unsigned short)(u >> 16);
}

// ---------------------------------------------------------------------------
// Kernel 1: offset conv. One thread per (b,h,w); 18 accumulators.
// Offset-conv weights staged in LDS transposed to [c*9+t][18] so the inner
// 18-FMA loop reads consecutive floats (wide ds loads, uniform -> broadcast).
// ---------------------------------------------------------------------------
__global__ __launch_bounds__(256) void dcn_offset_conv(
    const float* __restrict__ x, const float* __restrict__ w_off,
    const float* __restrict__ b_off, float* __restrict__ off) {
  __shared__ float wl[10368];  // 18 * 576, transposed
  const int tid = threadIdx.x;
  for (int i = tid; i < 10368; i += 256) {
    int oc = i / 576, r = i - oc * 576;   // coalesced read of w_off
    wl[r * 18 + oc] = w_off[i];
  }
  __syncthreads();

  const int pix = blockIdx.x * 256 + tid;
  const int w = pix & (Wn - 1);
  const int h = (pix >> 7) & (Hn - 1);
  const int b = pix >> 14;

  float acc[18];
#pragma unroll
  for (int oc = 0; oc < 18; ++oc) acc[oc] = b_off[oc];

  int lin[9];
#pragma unroll
  for (int t = 0; t < 9; ++t) {
    int yy = h + t / 3 - 1, xx = w + t % 3 - 1;
    bool v = ((unsigned)yy < (unsigned)Hn) && ((unsigned)xx < (unsigned)Wn);
    lin[t] = v ? yy * Wn + xx : -1;
  }

  const float* xb = x + (size_t)b * Cn * HWn;
  for (int c = 0; c < Cn; ++c) {
    const float* xc = xb + (size_t)c * HWn;
#pragma unroll
    for (int t = 0; t < 9; ++t) {
      float v = (lin[t] >= 0) ? xc[lin[t]] : 0.f;   // EXEC-masked load
      const float* wp = &wl[(c * 9 + t) * 18];
#pragma unroll
      for (int oc = 0; oc < 18; ++oc) acc[oc] = fmaf(v, wp[oc], acc[oc]);
    }
  }

  size_t obase = (size_t)b * 18 * HWn + (size_t)h * Wn + w;
#pragma unroll
  for (int oc = 0; oc < 18; ++oc) off[obase + (size_t)oc * HWn] = acc[oc];
}

// ---------------------------------------------------------------------------
// Kernel 2: sampling + WMMA GEMM.
// Block = 256 threads (8 wave32). Tile = 32 pixels (one row segment) x 64 oc.
// Wave (wid>>2, wid&3) -> (16-pixel M sub-tile, 16-channel N sub-tile).
// K = 576 processed as 2 halves (c in [0,32),[32,64)) of 9 chunks (K=32) each.
// LDS: offsets 2.3KB + A frags 18KB + B frags 36KB = 56.3KB -> ~5 blocks/WGP.
// ---------------------------------------------------------------------------
__global__ __launch_bounds__(256) void dcn_main(
    const float* __restrict__ x, const float* __restrict__ off,
    const float* __restrict__ wgt, float* __restrict__ out) {
  __shared__ float off_py[288];
  __shared__ float off_px[288];
  __shared__ __align__(32) unsigned short A_sh[2 * 9 * 32 * 16];  // 18432 B
  __shared__ __align__(32) unsigned short B_sh[4 * 9 * 32 * 16];  // 36864 B

  const int tid = threadIdx.x;
  const int wid = tid >> 5;
  const int lane = tid & 31;
  const int wb = blockIdx.x * 32;
  const int h = blockIdx.y;
  const int b = blockIdx.z;

  // --- sample coordinates for 32 pixels x 9 taps ---
  for (int pt = tid; pt < 288; pt += 256) {
    int p = pt / 9, k = pt - p * 9;
    int w0 = wb + p;
    size_t base = ((size_t)b * 18 + 2 * k) * HWn + (size_t)h * Wn + w0;
    float dy = off[base];
    float dx = off[base + HWn];
    off_py[pt] = (float)h + (float)(k / 3) - 1.f + dy;
    off_px[pt] = (float)w0 + (float)(k % 3) - 1.f + dx;
  }

  const int mt = wid >> 2;  // pixel sub-tile (0..1)
  const int nt = wid & 3;   // output-channel tile (0..3)
  v8f accv = {};

  const float* xb = x + (size_t)b * Cn * HWn;

  for (int ph = 0; ph < 2; ++ph) {
    __syncthreads();

    // --- B fragments (32x16 bf16, K-row striped across lane halves) ---
    for (int i = tid; i < 288 * 64; i += 256) {
      int o = i / 288;
      int kkr = i - o * 288;                       // kk within this half
      float wv = wgt[(size_t)o * KK + ph * 288 + kkr];  // coalesced
      int chunk = kkr >> 5, kc = kkr & 31;
      int ln = (o & 15) | (kc & 16);               // lanes 16-31 hold K>=16
      int j = (kc & 15) >> 1, hl = kc & 1;
      B_sh[(((o >> 4) * 9 + chunk) * 32 + ln) * 16 + j * 2 + hl] = f2bf(wv);
    }

    // --- A fragments via bilinear sampling (c in [32*ph, 32*ph+32)) ---
    for (int pt = tid; pt < 288; pt += 256) {
      int p = pt / 9, kt = pt - p * 9;
      float py = off_py[pt], px = off_px[pt];
      float y0f = floorf(py), x0f = floorf(px);
      float fy = py - y0f, fx = px - x0f;
      int iy0 = (int)y0f, ix0 = (int)x0f;
      int iy1 = iy0 + 1, ix1 = ix0 + 1;
      bool vy0 = (unsigned)iy0 < (unsigned)Hn;
      bool vy1 = (unsigned)iy1 < (unsigned)Hn;
      bool vx0 = (unsigned)ix0 < (unsigned)Wn;
      bool vx1 = (unsigned)ix1 < (unsigned)Wn;
      float wy0 = 1.f - fy, wx0 = 1.f - fx;
      float w00 = (vy0 && vx0) ? wy0 * wx0 : 0.f;
      float w01 = (vy0 && vx1) ? wy0 * fx : 0.f;
      float w10 = (vy1 && vx0) ? fy * wx0 : 0.f;
      float w11 = (vy1 && vx1) ? fy * fx : 0.f;
      int l00 = (vy0 && vx0) ? iy0 * Wn + ix0 : 0;
      int l01 = (vy0 && vx1) ? iy0 * Wn + ix1 : 0;
      int l10 = (vy1 && vx0) ? iy1 * Wn + ix0 : 0;
      int l11 = (vy1 && vx1) ? iy1 * Wn + ix1 : 0;

      const float* xc = xb + (size_t)(ph * 32) * HWn;
      const int mrow = p & 15, mtl = p >> 4;
#pragma unroll 8
      for (int ci = 0; ci < 32; ++ci) {
        const float* xp = xc + (size_t)ci * HWn;
        float v = xp[l00] * w00 + xp[l01] * w01 + xp[l10] * w10 + xp[l11] * w11;
        int kkr = ci * 9 + kt;
        int chunk = kkr >> 5, kc = kkr & 31;
        // A 16x32 layout used here (contiguous K, same as B): lanes 0-15 hold K in 0-15,
        // lanes 16-31 hold K in 16-31; 2 K per dword (lo/hi half).
        int ln = mrow | (kc & 16);
        int j = (kc & 15) >> 1;
        int hl = kc & 1;
        A_sh[((mtl * 9 + chunk) * 32 + ln) * 16 + j * 2 + hl] = f2bf(v);
      }
    }
    __syncthreads();

    // --- 9 WMMAs for this K half; fragments are 32B/lane contiguous rows ---
#pragma unroll
    for (int ch = 0; ch < 9; ++ch) {
      v16bf av = *reinterpret_cast<const v16bf*>(
          &A_sh[((mt * 9 + ch) * 32 + lane) * 16]);
      v16bf bv = *reinterpret_cast<const v16bf*>(
          &B_sh[((nt * 9 + ch) * 32 + lane) * 16]);
      accv = __builtin_amdgcn_wmma_f32_16x16x32_bf16(
          false, av, false, bv, (short)0, accv, false, false);
    }
  }

  __syncthreads();
  // --- stage C tile (16x16 f32) through LDS for coalesced global stores ---
  float* cs = reinterpret_cast<float*>(A_sh) + wid * 16 * 17;  // padded rows
  {
    int M0 = (lane < 16) ? 0 : 8;   // C/D layout: lanes 16-31 hold M = r+8
    int N = lane & 15;
#pragma unroll
    for (int r = 0; r < 8; ++r) cs[(M0 + r) * 17 + N] = accv[r];
  }
  __syncthreads();
  {
    int wtile = wb + mt * 16;
    int obase = nt * 16;
    int wpos = lane & 15;
#pragma unroll
    for (int r = 0; r < 8; ++r) {
      int o = 2 * r + (lane >> 4);           // 2 channel-rows per iteration
      float v = cs[wpos * 17 + o];           // conflict-free (17-padded)
      out[((size_t)(b * On + obase + o) * Hn + h) * Wn + wtile + wpos] = v;
    }
  }
}

// ---------------------------------------------------------------------------
extern "C" void kernel_launch(void* const* d_in, const int* in_sizes, int n_in,
                              void* d_out, int out_size, void* d_ws,
                              size_t ws_size, hipStream_t stream) {
  (void)in_sizes; (void)n_in; (void)out_size; (void)ws_size;
  const float* x     = (const float*)d_in[0];
  const float* w_off = (const float*)d_in[1];
  const float* b_off = (const float*)d_in[2];
  const float* wgt   = (const float*)d_in[3];
  float* out = (float*)d_out;
  float* off = (float*)d_ws;  // 4*18*128*128 floats = 4.72 MB scratch

  dcn_offset_conv<<<dim3((Bn * Hn * Wn) / 256), dim3(256), 0, stream>>>(
      x, w_off, b_off, off);
  dcn_main<<<dim3(Wn / 32, Hn, Bn), dim3(256), 0, stream>>>(x, off, wgt, out);
}
